// LSTM_52982716563922
// MI455X (gfx1250) — matrix operand order repaired and checked by reading, not compiled, not run on hardware
//
#include <hip/hip_runtime.h>

typedef __attribute__((ext_vector_type(16))) _Float16 v16h;
typedef __attribute__((ext_vector_type(8)))  _Float16 v8h;
typedef __attribute__((ext_vector_type(8)))  float    v8f;

// global (device-memory) address space versions for explicit global_load_*
typedef __attribute__((address_space(1))) const _Float16 gc_half;
typedef __attribute__((address_space(1))) const v16h     gc_v16h;

#define S_LEN 2048
#define BATCH 128
#define DIM   256     // D
#define HID   256     // H
#define GATES 1024    // 4H
#define OUTD  64
#define RS    528     // padded LDS row stride in halves (512 cols + 16 pad)

// Workgroup split-barrier that only drains DScnt (LDS ordering), leaving the
// just-issued global B-fragment prefetch loads in flight across the barrier.
// (__syncthreads conservatively drains LOADcnt too, serializing the prefetch.)
#define WG_BARRIER_LDS()                                         \
    asm volatile("s_wait_dscnt 0x0\n\t"                          \
                 "s_barrier_signal -1\n\t"                       \
                 "s_barrier_wait -1" ::: "memory")

// ---------------------------------------------------------------------------
// Kernel 0: pack [Wi; Wh] (f32, each [256,1024] row-major) into WMMA B-fragment
// order as f16. Fragment layout: Bp[((ct*16 + kb)*32 + lane)*16 + e] where
//   n = ct*16 + (lane&15),  k = kb*32 + (lane>>4)*8 + (e&7) + (e>>3)*16
// so each lane's 16-half B fragment is 32 contiguous bytes.
// ---------------------------------------------------------------------------
__global__ void pack_weights(const float* __restrict__ Wi,
                             const float* __restrict__ Wh,
                             _Float16* __restrict__ Bp) {
    int tid  = blockIdx.x * blockDim.x + threadIdx.x;   // 0 .. 524287
    int e    = tid & 15;
    int lane = (tid >> 4) & 31;
    int kb   = (tid >> 9) & 15;
    int ct   = tid >> 13;                               // 0..63
    int n = ct * 16 + (lane & 15);
    int k = kb * 32 + ((lane >> 4) << 3) + (e & 7) + ((e >> 3) << 4); // 0..511
    float w = (k < DIM) ? Wi[k * GATES + n] : Wh[(k - DIM) * GATES + n];
    Bp[tid] = (_Float16)w;
}

// ---------------------------------------------------------------------------
// Kernel 1: full LSTM recurrence. grid = 8 blocks (16 batch rows each),
// 512 threads = 16 waves. Wave t owns h-columns [16t, 16t+16) i.e. gate
// column-tiles {t, t+16, t+32, t+48}; cell state c kept in registers.
// Per step: gates = [x_t | h] (16x512, f16 in LDS) @ Bpack  (f32 accum WMMA).
// B fragments are software-pipelined one k-block ahead; the first fragment
// set is issued before the barrier so L2 latency overlaps the barrier wait.
// ---------------------------------------------------------------------------
__global__ void __launch_bounds__(512)
lstm_recurrence(const float* __restrict__ x,
                const float* __restrict__ h0,
                const float* __restrict__ Wo,
                const float* __restrict__ bi,
                const float* __restrict__ bo,
                const _Float16* __restrict__ Bp,
                float* __restrict__ out) {
    __shared__ _Float16 Ah[16 * RS];    // [16 rows][512 cols + pad]: x | h
    __shared__ float    hsm[16 * 260];  // final h (f32) for the output head
    __shared__ float    lg[16 * 64];    // logits

    const int tid  = threadIdx.x;
    const int wid  = tid >> 5;          // 0..15 (wave id == h column tile)
    const int lane = tid & 31;
    const int nl   = lane & 15;
    const int hb   = lane >> 4;
    const int r0   = blockIdx.x * 16;   // first batch row owned by this block

    // x/h loader mapping: each thread handles 8 consecutive cols of one row
    const int lm  = tid >> 5;           // row 0..15
    const int ld0 = (tid & 31) * 8;     // col 0..248

    // ---- init A h-part (cols 256..511) from h0, c0 = h0 (reference quirk) --
    {
        const float* hp = h0 + (size_t)(r0 + lm) * HID + ld0;
        v8h hv;
        for (int j = 0; j < 8; ++j) hv[j] = (_Float16)hp[j];
        *(v8h*)&Ah[lm * RS + 256 + ld0] = hv;
    }
    v8f creg;
    for (int v = 0; v < 8; ++v)
        creg[v] = h0[(size_t)(r0 + v + 8 * hb) * HID + wid * 16 + nl];

    // ---- per-lane biases for the 4 gate tiles this wave owns ---------------
    const float bias_i = bi[0 * HID + wid * 16 + nl];
    const float bias_f = bi[1 * HID + wid * 16 + nl];
    const float bias_g = bi[2 * HID + wid * 16 + nl];
    const float bias_o = bi[3 * HID + wid * 16 + nl];

    // ---- per-lane B fragment offsets (in halves) ---------------------------
    const size_t oi = (size_t)(wid     ) * 8192 + lane * 16;
    const size_t of = (size_t)(wid + 16) * 8192 + lane * 16;
    const size_t og = (size_t)(wid + 32) * 8192 + lane * 16;
    const size_t oo = (size_t)(wid + 48) * 8192 + lane * 16;

    // A fragment base: lane reads row nl; chunks at kb*32 + hb*8 (+16)
    const int aoff = nl * RS + hb * 8;

    const float* xrow = x + (size_t)(r0 + lm) * DIM + ld0;
    const int hcol = wid * 16 + nl;

    // Weight base pointer in explicit global address space (so loads lower to
    // global_load_b128, not flat_load). Laundered through an empty asm each
    // step so LICM cannot hoist 2KB/lane of B-fragment loads out of the loop
    // (which previously spilled to scratch). Re-read from L2 every step.
    gc_half* Bs = (gc_half*)Bp;

    for (int s = 0; s < S_LEN; ++s) {
        // ---- stream x_t tile into A (f32 -> f16), prefetch next step ------
        {
            const float4 f0 = *(const float4*)(xrow);
            const float4 f1 = *(const float4*)(xrow + 4);
            v8h hv;
            hv[0] = (_Float16)f0.x; hv[1] = (_Float16)f0.y;
            hv[2] = (_Float16)f0.z; hv[3] = (_Float16)f0.w;
            hv[4] = (_Float16)f1.x; hv[5] = (_Float16)f1.y;
            hv[6] = (_Float16)f1.z; hv[7] = (_Float16)f1.w;
            *(v8h*)&Ah[lm * RS + ld0] = hv;
            __builtin_prefetch(xrow + BATCH * DIM, 0, 1);
            xrow += BATCH * DIM;
        }

        // defeat loop-invariant hoisting of the weight loads (uniform -> SGPR)
        asm volatile("" : "+s"(Bs));

        // issue kb=0 B fragments BEFORE the barrier: L2 latency overlaps the
        // barrier wait (these depend only on Bs, not on LDS contents)
        v16h pbI = *(gc_v16h*)(Bs + oi);
        v16h pbF = *(gc_v16h*)(Bs + of);
        v16h pbG = *(gc_v16h*)(Bs + og);
        v16h pbO = *(gc_v16h*)(Bs + oo);

        WG_BARRIER_LDS();   // drain DScnt only; B loads stay in flight

        // ---- gates = [x_t | h] @ [Wi; Wh] + bias, f32 accumulate ----------
        v8f acc_i, acc_f, acc_g, acc_o;
        for (int v = 0; v < 8; ++v) {
            acc_i[v] = bias_i; acc_f[v] = bias_f;
            acc_g[v] = bias_g; acc_o[v] = bias_o;
        }
        #pragma unroll
        for (int kb = 0; kb < 16; ++kb) {
            v16h bI = pbI, bF = pbF, bG = pbG, bO = pbO;
            if (kb < 15) {   // pipeline: fetch next k-block during these WMMAs
                pbI = *(gc_v16h*)(Bs + oi + (kb + 1) * 512);
                pbF = *(gc_v16h*)(Bs + of + (kb + 1) * 512);
                pbG = *(gc_v16h*)(Bs + og + (kb + 1) * 512);
                pbO = *(gc_v16h*)(Bs + oo + (kb + 1) * 512);
            }
            v8h a0 = *(const v8h*)&Ah[aoff + kb * 32];
            v8h a1 = *(const v8h*)&Ah[aoff + kb * 32 + 16];
            v16h a = __builtin_shufflevector(a0, a1,
                        0,1,2,3,4,5,6,7,8,9,10,11,12,13,14,15);
            acc_i = __builtin_amdgcn_wmma_f32_16x16x32_f16(false, a, false, bI,
                        (short)0, acc_i, false, false);
            acc_f = __builtin_amdgcn_wmma_f32_16x16x32_f16(false, a, false, bF,
                        (short)0, acc_f, false, false);
            acc_g = __builtin_amdgcn_wmma_f32_16x16x32_f16(false, a, false, bG,
                        (short)0, acc_g, false, false);
            acc_o = __builtin_amdgcn_wmma_f32_16x16x32_f16(false, a, false, bO,
                        (short)0, acc_o, false, false);
        }

        WG_BARRIER_LDS();   // all GEMM reads of A done before h is rewritten

        // ---- elementwise LSTM cell update, all in f32 registers ------------
        for (int v = 0; v < 8; ++v) {
            float ig = 1.0f / (1.0f + __expf(-acc_i[v]));
            float fg = 1.0f / (1.0f + __expf(-acc_f[v]));
            float t2 = __expf(-2.0f * acc_g[v]);
            float gg = (1.0f - t2) / (1.0f + t2);          // tanh
            float oq = 1.0f / (1.0f + __expf(-acc_o[v]));
            float c  = creg[v] * fg + ig * gg;
            creg[v]  = c;
            float tc = __expf(-2.0f * c);
            float h  = oq * (1.0f - tc) / (1.0f + tc);
            int m = v + 8 * hb;
            Ah[m * RS + 256 + hcol] = (_Float16)h;          // h for next step
            if (s == S_LEN - 1) {
                hsm[m * 260 + hcol] = h;
                out[8192 + (size_t)(r0 + m) * HID + hcol] = h;  // h output
            }
        }
    }
    __syncthreads();

    // ---- logits = h @ Wo + bo  (tiny: 16x64, f32 VALU) ---------------------
    for (int idx = tid; idx < 16 * OUTD; idx += 512) {
        int m = idx >> 6, n = idx & 63;
        float acc = bo[n];
        for (int k = 0; k < HID; ++k)
            acc += hsm[m * 260 + k] * Wo[k * OUTD + n];
        lg[m * 64 + n] = acc;
    }
    __syncthreads();

    // ---- softmax over 64 logits; wave `wid` handles row `wid` --------------
    {
        int row = wid;
        float v0 = lg[row * 64 + lane];
        float v1 = lg[row * 64 + lane + 32];
        float mx = fmaxf(v0, v1);
        for (int off = 16; off > 0; off >>= 1)
            mx = fmaxf(mx, __shfl_xor(mx, off, 32));
        float e0 = __expf(v0 - mx), e1 = __expf(v1 - mx);
        float sm = e0 + e1;
        for (int off = 16; off > 0; off >>= 1)
            sm += __shfl_xor(sm, off, 32);
        out[(size_t)(r0 + row) * 64 + lane]      = e0 / sm;
        out[(size_t)(r0 + row) * 64 + lane + 32] = e1 / sm;
    }
}

// ---------------------------------------------------------------------------
extern "C" void kernel_launch(void* const* d_in, const int* in_sizes, int n_in,
                              void* d_out, int out_size, void* d_ws, size_t ws_size,
                              hipStream_t stream) {
    const float* x  = (const float*)d_in[0];   // [2048,128,256]
    const float* h0 = (const float*)d_in[1];   // [128,256]
    const float* Wi = (const float*)d_in[2];   // [256,1024]
    const float* Wh = (const float*)d_in[3];   // [256,1024]
    const float* Wo = (const float*)d_in[4];   // [256,64]
    const float* bi = (const float*)d_in[5];   // [1024]
    const float* bo = (const float*)d_in[6];   // [64]
    _Float16* Bp = (_Float16*)d_ws;            // 1 MiB packed f16 weights
    float* out = (float*)d_out;                // [128*64 softmax][128*256 h]

    pack_weights<<<2048, 256, 0, stream>>>(Wi, Wh, Bp);
    lstm_recurrence<<<8, 512, 0, stream>>>(x, h0, Wo, bi, bo, Bp, out);
}
